// prj_module_16054587752954
// MI455X (gfx1250) — matrix operand order repaired and checked by reading, not compiled, not run on hardware
//
#include <hip/hip_runtime.h>

// out = x - w * A^T (A x - proj),  A: [11520 x 9216] fp32.
// HBM-bound on streaming A twice (848 MB -> ~36us @23.3TB/s).
// bf16 WMMA, fp32 accumulate. A converted on the fly with v_cvt_pk_bf16_f32;
// x and temp converted exactly once (bf16 copies live in d_ws).

typedef __attribute__((ext_vector_type(16))) __bf16 v16bf;
typedef __attribute__((ext_vector_type(8)))  float  v8f;

#define HW   9216     // 96*96 pixels
#define NVD  11520    // 120*96 sinogram bins
#define NB   64       // batch
#define LSTR 40       // LDS row stride (bf16 elems): 80B rows, 16B aligned, conflict-free

union FragBF { uint4 q[2]; v16bf v; };

// gfx1250 packed fp32->bf16 convert (RNE), 1 instruction per 2 elements.
static __device__ __forceinline__ unsigned int cvt_pk_bf16(float lo, float hi) {
  unsigned int r;
  asm("v_cvt_pk_bf16_f32 %0, %1, %2" : "=v"(r) : "v"(lo), "v"(hi));
  return r;
}
static __device__ __forceinline__ unsigned short cvt1_bf16(float v) {
  return (unsigned short)(cvt_pk_bf16(v, v) & 0xffffu);
}
static __device__ __forceinline__ uint4 pack8(float4 f0, float4 f1) {
  uint4 p;
  p.x = cvt_pk_bf16(f0.x, f0.y);
  p.y = cvt_pk_bf16(f0.z, f0.w);
  p.z = cvt_pk_bf16(f1.x, f1.y);
  p.w = cvt_pk_bf16(f1.z, f1.w);
  return p;
}

// ---- one-shot fp32 -> bf16 copy (x image: 589824 elems, 8 per thread) ----
__global__ __launch_bounds__(256)
void cvt_x_bf16(const float* __restrict__ x, unsigned short* __restrict__ xb) {
  const int i = (blockIdx.x * 256 + threadIdx.x) * 8;
  float4 f0 = *(const float4*)(x + i);
  float4 f1 = *(const float4*)(x + i + 4);
  *(uint4*)(xb + i) = pack8(f0, f1);
}

// ---------------- Stage 1: tempb[b,n] = bf16( sum_p A[n,p]*x[b,p] - proj[b,n] ) ----------------
// GEMM M=64 (batch), N=11520 (n), K=9216 (p). A-frag = x tile (bf16 copy), B-frag = A[n,p] tile.
__global__ __launch_bounds__(256)
void sart_stage1_residual(const unsigned short* __restrict__ xb,
                          const float* __restrict__ proj,
                          const float* __restrict__ A,
                          unsigned short* __restrict__ tempb) {
  __shared__ __align__(16) unsigned short lds_x[NB * LSTR];
  __shared__ __align__(16) unsigned short lds_a[NB * LSTR];

  const int t    = threadIdx.x;
  const int lane = t & 31;
  const int w    = t >> 5;
  const int h    = lane >> 4;
  const int l16  = lane & 15;
  const int ntile = w & 3;
  const int mt0   = (w >> 2) * 2;
  const int n0    = blockIdx.x * 64;

  v8f acc0 = {}, acc1 = {};

  // fragment LDS offsets (ISA 16-bit WMMA layouts)
  const int aoff0 = (mt0 * 16 + l16) * LSTR + h * 8;
  const int aoff1 = ((mt0 + 1) * 16 + l16) * LSTR + h * 8;
  const int boff  = (ntile * 16 + l16) * LSTR + h * 8;

  // fill coords: each thread owns one 8-elem row segment of each 64x32 tile
  const int frow = t >> 2;
  const int fseg = (t & 3) * 8;
  const unsigned short* gx = xb + (long)frow * HW + fseg;
  const float*          ga = A + (long)(n0 + frow) * HW + fseg;

  // prologue: tile k0 = 0 into registers
  uint4  rx  = *(const uint4*)gx;
  float4 ra0 = *(const float4*)ga;
  float4 ra1 = *(const float4*)(ga + 4);

  for (int k0 = 0; k0 < HW; k0 += 32) {
    *(uint4*)(lds_x + frow * LSTR + fseg) = rx;             // pure copy (pre-converted)
    *(uint4*)(lds_a + frow * LSTR + fseg) = pack8(ra0, ra1); // 4x v_cvt_pk_bf16_f32
    __syncthreads();

    if (k0 + 32 < HW) {  // software pipeline: issue next tile's loads now
      rx  = *(const uint4*)(gx + k0 + 32);
      ra0 = *(const float4*)(ga + k0 + 32);
      ra1 = *(const float4*)(ga + k0 + 36);
      __builtin_prefetch(ga + k0 + 64, 0, 0);
    }

    FragBF a0, a1, b;
    a0.q[0] = *(const uint4*)(lds_x + aoff0);
    a0.q[1] = *(const uint4*)(lds_x + aoff0 + 16);
    a1.q[0] = *(const uint4*)(lds_x + aoff1);
    a1.q[1] = *(const uint4*)(lds_x + aoff1 + 16);
    b.q[0]  = *(const uint4*)(lds_a + boff);
    b.q[1]  = *(const uint4*)(lds_a + boff + 16);

    acc0 = __builtin_amdgcn_wmma_f32_16x16x32_bf16(false, a0.v, false, b.v, (short)0, acc0, false, false);
    acc1 = __builtin_amdgcn_wmma_f32_16x16x32_bf16(false, a1.v, false, b.v, (short)0, acc1, false, false);
    __syncthreads();
  }

  const int n = n0 + ntile * 16 + l16;
#pragma unroll
  for (int r = 0; r < 8; ++r) {
    const int b0 = mt0 * 16 + r + 8 * h;
    const int b1 = (mt0 + 1) * 16 + r + 8 * h;
    tempb[(long)b0 * NVD + n] = cvt1_bf16(acc0[r] - proj[(long)b0 * NVD + n]);
    tempb[(long)b1 * NVD + n] = cvt1_bf16(acc1[r] - proj[(long)b1 * NVD + n]);
  }
}

// ---------------- Stage 2: out[b,p] = x[b,p] - w * sum_n A[n,p]*temp[b,n] ----------------
// GEMM M=64 (batch), N=9216 (p), K=11520 (n). B-frag = A[n,p] with K=n -> staged transposed.
__global__ __launch_bounds__(256)
void sart_stage2_backproj(const unsigned short* __restrict__ tempb,
                          const float* __restrict__ A,
                          const float* __restrict__ x,
                          const float* __restrict__ weight,
                          float* __restrict__ out) {
  __shared__ __align__(16) unsigned short lds_t[NB * LSTR];
  __shared__ __align__(16) unsigned short lds_b[NB * LSTR];  // [p_local][n_local]

  const int t    = threadIdx.x;
  const int lane = t & 31;
  const int w    = t >> 5;
  const int h    = lane >> 4;
  const int l16  = lane & 15;
  const int ptile = w & 3;
  const int mt0   = (w >> 2) * 2;
  const int p0    = blockIdx.x * 64;

  v8f acc0 = {}, acc1 = {};

  const int aoff0 = (mt0 * 16 + l16) * LSTR + h * 8;
  const int aoff1 = ((mt0 + 1) * 16 + l16) * LSTR + h * 8;
  const int boff  = (ptile * 16 + l16) * LSTR + h * 8;

  const int frow = t >> 2;           // temp tile fill
  const int fseg = (t & 3) * 8;
  const unsigned short* gt = tempb + (long)frow * NVD + fseg;

  const int trow = t >> 3;           // A tile fill: n row 0..31
  const int tcol = (t & 7) * 8;      // p col group
  const float* ga = A + (long)trow * HW + p0 + tcol;

  uint4  rt  = *(const uint4*)gt;
  float4 rb0 = *(const float4*)ga;
  float4 rb1 = *(const float4*)(ga + 4);

  for (int k0 = 0; k0 < NVD; k0 += 32) {
    *(uint4*)(lds_t + frow * LSTR + fseg) = rt;
    {
      uint4 pk = pack8(rb0, rb1);
      unsigned int pw[4] = {pk.x, pk.y, pk.z, pk.w};
#pragma unroll
      for (int j = 0; j < 4; ++j) {   // transpose: K(n)-pairs contiguous per p row
        lds_b[(tcol + 2 * j    ) * LSTR + trow] = (unsigned short)(pw[j] & 0xffffu);
        lds_b[(tcol + 2 * j + 1) * LSTR + trow] = (unsigned short)(pw[j] >> 16);
      }
    }
    __syncthreads();

    if (k0 + 32 < NVD) {
      rt  = *(const uint4*)(gt + k0 + 32);
      rb0 = *(const float4*)(ga + (long)(k0 + 32) * HW);
      rb1 = *(const float4*)(ga + (long)(k0 + 32) * HW + 4);
      __builtin_prefetch(ga + (long)(k0 + 64) * HW, 0, 0);
    }

    FragBF a0, a1, b;
    a0.q[0] = *(const uint4*)(lds_t + aoff0);
    a0.q[1] = *(const uint4*)(lds_t + aoff0 + 16);
    a1.q[0] = *(const uint4*)(lds_t + aoff1);
    a1.q[1] = *(const uint4*)(lds_t + aoff1 + 16);
    b.q[0]  = *(const uint4*)(lds_b + boff);
    b.q[1]  = *(const uint4*)(lds_b + boff + 16);

    acc0 = __builtin_amdgcn_wmma_f32_16x16x32_bf16(false, a0.v, false, b.v, (short)0, acc0, false, false);
    acc1 = __builtin_amdgcn_wmma_f32_16x16x32_bf16(false, a1.v, false, b.v, (short)0, acc1, false, false);
    __syncthreads();
  }

  const float wg = weight[0];
  const int p = p0 + ptile * 16 + l16;
#pragma unroll
  for (int r = 0; r < 8; ++r) {
    const int b0 = mt0 * 16 + r + 8 * h;
    const int b1 = (mt0 + 1) * 16 + r + 8 * h;
    out[(long)b0 * HW + p] = x[(long)b0 * HW + p] - wg * acc0[r];
    out[(long)b1 * HW + p] = x[(long)b1 * HW + p] - wg * acc1[r];
  }
}

extern "C" void kernel_launch(void* const* d_in, const int* in_sizes, int n_in,
                              void* d_out, int out_size, void* d_ws, size_t ws_size,
                              hipStream_t stream) {
  (void)in_sizes; (void)n_in; (void)out_size; (void)ws_size;
  const float* x      = (const float*)d_in[0];   // [64,1,96,96]
  const float* proj   = (const float*)d_in[1];   // [64,1,120,96]
  const float* weight = (const float*)d_in[2];   // [1]
  const float* A      = (const float*)d_in[3];   // [11520,9216]
  float* out = (float*)d_out;

  unsigned short* tempb = (unsigned short*)d_ws;             // [64,11520] bf16 (1.47 MB)
  unsigned short* xb    = tempb + (size_t)NB * NVD;          // [64,9216]  bf16 (1.18 MB)

  cvt_x_bf16<<<(NB * HW) / (256 * 8), 256, 0, stream>>>(x, xb);
  sart_stage1_residual<<<NVD / 64, 256, 0, stream>>>(xb, proj, A, tempb);
  sart_stage2_backproj<<<HW / 64, 256, 0, stream>>>(tempb, A, x, weight, out);
}
